// HeLPFormer_27092653703585
// MI455X (gfx1250) — matrix operand order repaired and checked by reading, not compiled, not run on hardware
//
#include <hip/hip_runtime.h>
#include <stdint.h>

#define NN    8192
#define SQ    32
#define EMB   256
#define NHD   8
#define HD    32
#define NTY   4
#define NE    131072
#define ASTR  264   // half stride for f16 LDS tiles (16B-aligned rows: 264*2=528=33*16)
#define FSTR  264   // float stride for LDS fl/fr
#define SLOPE 0.2f
#define LNEPS 1e-5f

typedef _Float16 half_t;
typedef __attribute__((ext_vector_type(8)))  _Float16 v8h;
typedef __attribute__((ext_vector_type(16))) _Float16 v16h;
typedef __attribute__((ext_vector_type(8)))  float    v8f;

// ---------------- WMMA fragment helpers (CDNA5 16x16x32 f16 layouts) ----------------

// A: 16-bit 16x32 (MxK). lane l: row = l&15, half = l>>4;
// elems 0..7  -> K = kBase + half*8 + 0..7
// elems 8..15 -> K = kBase + 16 + half*8 + 0..7
__device__ __forceinline__ v16h load_a_frag(const half_t* As, int mBase, int kBase, int lane) {
  const half_t* p = As + (mBase + (lane & 15)) * ASTR + kBase + (lane >> 4) * 8;
  v8h lo = *(const v8h*)p;
  v8h hi = *(const v8h*)(p + 16);
  return __builtin_shufflevector(lo, hi, 0,1,2,3,4,5,6,7,8,9,10,11,12,13,14,15);
}

// B: 16-bit 32x16 (KxN). lane l: col = l&15; elems 0..7 -> K = kBase + (l>>4)*8 + 0..7,
// elems 8..15 -> K = kBase + 16 + (l>>4)*8 + 0..7 (same as A).
// Wt is the TRANSPOSED weight (row = output col n, 256 K contiguous) -> two 16B loads.
__device__ __forceinline__ v16h load_b_frag(const half_t* Wt, int nBase, int kBase, int lane) {
  const half_t* p = Wt + (size_t)(nBase + (lane & 15)) * EMB + kBase + (lane >> 4) * 8;
  v8h lo = *(const v8h*)p;
  v8h hi = *(const v8h*)(p + 16);
  return __builtin_shufflevector(lo, hi, 0,1,2,3,4,5,6,7,8,9,10,11,12,13,14,15);
}

// C[0:32,0:256] = A(32x256 f16 LDS tile) @ W  (W given transposed as Wt[n][k])
// 32 output tiles (2 Mtiles x 16 Ntiles), 8 waves -> 4 tiles/wave, K in 8 steps of 32.
__device__ __forceinline__ void gemm_32x256(const half_t* As, const half_t* Wt,
                                            float* C, int ldc, int tid) {
  int lane = tid & 31, wave = tid >> 5;
  for (int ti = 0; ti < 4; ++ti) {
    int t  = wave * 4 + ti;
    int mt = (t >> 4) * 16;
    int nt = (t & 15) * 16;
    v8f acc = {};
    #pragma unroll
    for (int k0 = 0; k0 < EMB; k0 += 32) {
      v16h a = load_a_frag(As, mt, k0, lane);
      v16h b = load_b_frag(Wt, nt, k0, lane);
      acc = __builtin_amdgcn_wmma_f32_16x16x32_f16(false, a, false, b, (short)0, acc, false, false);
    }
    // D layout: lane l -> col = nt + (l&15), rows = mt + (l>>4)*8 + i
    int rb = mt + (lane >> 4) * 8;
    int cc = nt + (lane & 15);
    #pragma unroll
    for (int i = 0; i < 8; ++i)
      C[(size_t)(rb + i) * ldc + cc] = acc[i];
  }
}

// ---------------- small kernels ----------------

__global__ __launch_bounds__(256) void k_fc(const float* feat, const float* W, const float* bias,
                                            float* outp, int rows, int indim) {
  int idx = blockIdx.x * 256 + threadIdx.x;
  if (idx >= rows * EMB) return;
  int r = idx >> 8, c = idx & 255;
  float acc = bias[c];
  const float* f = feat + (size_t)r * indim;
  for (int k = 0; k < indim; ++k) acc += f[k] * W[(size_t)k * EMB + c];
  outp[idx] = acc;
}

__global__ __launch_bounds__(256) void k_node_init(float* odeg, float* ideg, float* R,
                                                   const float* type_emb, const int* node_type) {
  int i = blockIdx.x * 256 + threadIdx.x;
  if (i >= NN) return;
  odeg[i] = 1.f; ideg[i] = 1.f;               // self-loop
  int t = node_type[i];
  #pragma unroll
  for (int j = 0; j < NTY; ++j) R[i * NTY + j] = type_emb[t * NTY + j];
}

__global__ __launch_bounds__(256) void k_deg_edges(const int* src, const int* dst,
                                                   float* odeg, float* ideg) {
  int e = blockIdx.x * 256 + threadIdx.x;
  if (e >= NE) return;
  atomicAdd(&odeg[src[e]], 1.f);
  atomicAdd(&ideg[dst[e]], 1.f);
}

__global__ __launch_bounds__(256) void k_deg_fin(float* d) {
  int i = blockIdx.x * 256 + threadIdx.x;
  if (i < NN) d[i] = rsqrtf(fmaxf(d[i], 1.f));
}

// transpose + convert a 256x256 fp32 weight into f16: Wt[n*256+k] = (half)W[k*256+n]
__global__ __launch_bounds__(256) void k_w2h_t(const float* W, half_t* Wt) {
  int idx = blockIdx.x * 256 + threadIdx.x;   // 65536
  int n = idx >> 8, k = idx & 255;
  Wt[idx] = (half_t)W[(size_t)k * EMB + n];
}

__global__ __launch_bounds__(256) void k_zero(float* p, int n) {
  int i = blockIdx.x * 256 + threadIdx.x;
  if (i < n) p[i] = 0.f;
}

// x = (gh * n_out) @ Wgcn  via WMMA
__global__ __launch_bounds__(256) void k_gemm_scaled(const float* A, const float* rowScale,
                                                     const half_t* Wt, float* C) {
  __shared__ half_t As[32 * ASTR];
  int m0 = blockIdx.x * 32, tid = threadIdx.x;
  for (int idx = tid; idx < 32 * EMB; idx += 256) {
    int r = idx >> 8, c = idx & 255;
    As[r * ASTR + c] = (half_t)(A[(size_t)(m0 + r) * EMB + c] * rowScale[m0 + r]);
  }
  __syncthreads();
  gemm_32x256(As, Wt, C + (size_t)m0 * EMB, EMB, tid);
}

// accum[dst[e]] += x[src[e]]  (one block per edge, thread = channel)
__global__ __launch_bounds__(256) void k_scatter_emb(const float* x, const int* src, const int* dst,
                                                     float* accum) {
  int e = blockIdx.x, c = threadIdx.x;
  atomicAdd(&accum[(size_t)dst[e] * EMB + c], x[(size_t)src[e] * EMB + c]);
}

// gh = relu((edgeSum + selfLoop x) * n_in + b)
__global__ __launch_bounds__(256) void k_gcn_fin(float* gh, const float* x, const float* nin,
                                                 const float* b) {
  int idx = blockIdx.x * 256 + threadIdx.x;
  if (idx >= NN * EMB) return;
  int i = idx >> 8, c = idx & 255;
  float v = (gh[idx] + x[idx]) * nin[i] + b[c];
  gh[idx] = v > 0.f ? v : 0.f;
}

__global__ __launch_bounds__(256) void k_re_msg(const float* R, const float* nout, const float* Wre,
                                                const float* wtre, const int* node_type, float* Rt) {
  int idx = blockIdx.x * 256 + threadIdx.x;
  if (idx >= NN * NTY) return;
  int i = idx >> 2, t = idx & 3;
  float s = nout[i], acc = 0.f;
  #pragma unroll
  for (int u = 0; u < NTY; ++u) acc += R[i * NTY + u] * s * Wre[u * NTY + t];
  Rt[idx] = acc * wtre[node_type[i]];
}

__global__ __launch_bounds__(256) void k_re_scatter(const float* Rt, const int* src, const int* dst,
                                                    float* Racc) {
  int idx = blockIdx.x * 256 + threadIdx.x;
  if (idx >= NE * NTY) return;
  int e = idx >> 2, t = idx & 3;
  atomicAdd(&Racc[(size_t)dst[e] * NTY + t], Rt[(size_t)src[e] * NTY + t]);
}

__global__ __launch_bounds__(256) void k_re_fin(float* R, const float* Rt, const float* nin,
                                                const float* b) {
  int idx = blockIdx.x * 256 + threadIdx.x;
  if (idx >= NN * NTY) return;
  int i = idx >> 2, t = idx & 3;
  float v = (R[idx] + Rt[idx]) * nin[i] + b[t];
  R[idx] = v > 0.f ? v : 0.f;
}

__global__ __launch_bounds__(256) void k_l2n_out(const float* gh, float* out) {
  __shared__ float red[8];
  int i = blockIdx.x, tid = threadIdx.x;
  float v = gh[(size_t)i * EMB + tid];
  float s = v * v;
  for (int off = 16; off > 0; off >>= 1) s += __shfl_xor(s, off, 32);
  if ((tid & 31) == 0) red[tid >> 5] = s;
  __syncthreads();
  if (tid == 0) {
    float t = 0.f;
    for (int w = 0; w < 8; ++w) t += red[w];
    red[0] = 1.f / fmaxf(sqrtf(t), 1e-12f);
  }
  __syncthreads();
  out[(size_t)i * 768 + tid] = v * red[0];
}

__global__ __launch_bounds__(256) void k_gather(const float* gh, const int* seqs, float* h) {
  size_t idx = (size_t)blockIdx.x * 256 + threadIdx.x;   // NN*SQ*EMB
  int bs = (int)(idx >> 8), c = (int)(idx & 255);
  h[idx] = gh[(size_t)seqs[bs] * EMB + c];
}

// ---------------- attention layer: one node per block, wave = head ----------------

__global__ __launch_bounds__(256) void k_attn(float* __restrict__ h, const float* __restrict__ R,
                                              const int* __restrict__ seqs,
                                              const half_t* Wl_t, const half_t* Wr_t, const half_t* Wfin_t,
                                              const float* al, const float* ar,
                                              const float* Wrs, const float* Wrt,
                                              const float* g, const float* bb,
                                              float* __restrict__ out, int outColOff) {
  extern __shared__ char smem[];
  half_t* As = (half_t*)smem;                                   // 32*ASTR halves (16896 B)
  float* fl = (float*)(smem + 32 * ASTR * sizeof(half_t));      // 32*FSTR  (reused as h_sa)
  float* fr = fl + 32 * FSTR;                                   // 32*FSTR  (reused as h_out)
  float* sc = fr + 32 * FSTR;                                   // 8*32*32
  float* rk = sc + NHD * SQ * SQ;                               // 32*32
  float* rq = rk + SQ * NHD * NTY;                              // 32*32

  int b = blockIdx.x, tid = threadIdx.x;
  int lane = tid & 31, wave = tid >> 5;
  float* hb = h + (size_t)b * SQ * EMB;

  // h tile -> f16 LDS
  for (int idx = tid; idx < SQ * EMB; idx += 256) {
    int r = idx >> 8, c = idx & 255;
    As[r * ASTR + c] = (half_t)hb[idx];
  }
  // rk[s][h*4+r] = rh @ Wrs ; rq likewise with Wrt
  #pragma unroll
  for (int q = 0; q < 4; ++q) {
    int lin = tid * 4 + q;            // 0..1023
    int s = lin >> 5, col = lin & 31;
    const float* Rr = R + (size_t)seqs[b * SQ + s] * NTY;
    float a0 = 0.f, a1 = 0.f;
    #pragma unroll
    for (int t = 0; t < NTY; ++t) {
      float rv = Rr[t];
      a0 += rv * Wrs[t * 32 + col];
      a1 += rv * Wrt[t * 32 + col];
    }
    rk[lin] = a0; rq[lin] = a1;
  }
  __syncthreads();

  gemm_32x256(As, Wl_t, fl, FSTR, tid);   // fl = h @ Wl
  gemm_32x256(As, Wr_t, fr, FSTR, tid);   // fr = h @ Wr
  __syncthreads();

  // per-head attention: wave = head, lane = seq position (scores) / channel (ctx)
  {
    int hh = wave;
    float sl = 0.f, sr = 0.f;
    #pragma unroll
    for (int d = 0; d < HD; ++d) {
      float vl = fl[lane * FSTR + hh * HD + d];
      float vr = fr[lane * FSTR + hh * HD + d];
      vl = vl > 0.f ? vl : SLOPE * vl;
      vr = vr > 0.f ? vr : SLOPE * vr;
      sl += vl * al[d];
      sr += vr * ar[d];
    }
    float* scw = sc + hh * SQ * SQ;
    for (int i = 0; i < SQ; ++i) {
      float sli = __shfl(sl, i, 32);
      float dotr = 0.f;
      #pragma unroll
      for (int r = 0; r < NTY; ++r)
        dotr += rk[i * 32 + hh * NTY + r] * rq[lane * 32 + hh * NTY + r];
      float s = sli + sr + dotr;                      // BETA=1, TEMPER=1
      float m = s;
      for (int off = 16; off > 0; off >>= 1) m = fmaxf(m, __shfl_xor(m, off, 32));
      float e = __expf(s - m);
      float ssum = e;
      for (int off = 16; off > 0; off >>= 1) ssum += __shfl_xor(ssum, off, 32);
      scw[i * SQ + lane] = e / ssum;
    }
    // ctx = attn @ fr ; write into this head's column slice of fl (h_sa)
    for (int i = 0; i < SQ; ++i) {
      float acc = 0.f;
      for (int j = 0; j < SQ; ++j)
        acc += scw[i * SQ + j] * fr[j * FSTR + hh * HD + lane];
      fl[i * FSTR + hh * HD + lane] = acc;
    }
  }
  __syncthreads();

  // h_sa -> f16, ho = h_sa @ Wfin (into fr)
  for (int idx = tid; idx < SQ * EMB; idx += 256) {
    int r = idx >> 8, c = idx & 255;
    As[r * ASTR + c] = (half_t)fl[r * FSTR + c];
  }
  __syncthreads();
  gemm_32x256(As, Wfin_t, fr, FSTR, tid);
  __syncthreads();

  // residual + LayerNorm, in-place h update; row 0 -> l2n -> out slab
  for (int rr = 0; rr < 4; ++rr) {
    int r = wave + rr * 8;
    float x[8], s1 = 0.f, s2 = 0.f;
    #pragma unroll
    for (int q = 0; q < 8; ++q) {
      int c = lane + q * 32;
      float v = hb[r * EMB + c] + fr[r * FSTR + c];
      x[q] = v; s1 += v; s2 += v * v;
    }
    for (int off = 16; off > 0; off >>= 1) { s1 += __shfl_xor(s1, off, 32); s2 += __shfl_xor(s2, off, 32); }
    float mean = s1 * (1.f / EMB);
    float var  = s2 * (1.f / EMB) - mean * mean;
    float rstd = rsqrtf(var + LNEPS);
    float y[8];
    #pragma unroll
    for (int q = 0; q < 8; ++q) {
      int c = lane + q * 32;
      y[q] = (x[q] - mean) * rstd * g[c] + bb[c];
      hb[r * EMB + c] = y[q];
    }
    if (wave == 0 && rr == 0) {   // row 0: l2-normalize into output
      float n2 = 0.f;
      #pragma unroll
      for (int q = 0; q < 8; ++q) n2 += y[q] * y[q];
      for (int off = 16; off > 0; off >>= 1) n2 += __shfl_xor(n2, off, 32);
      float inv = 1.f / fmaxf(sqrtf(n2), 1e-12f);
      #pragma unroll
      for (int q = 0; q < 8; ++q) {
        int c = lane + q * 32;
        out[(size_t)b * 768 + outColOff + c] = y[q] * inv;
      }
    }
  }
}

// ---------------- host ----------------

extern "C" void kernel_launch(void* const* d_in, const int* in_sizes, int n_in,
                              void* d_out, int out_size, void* d_ws, size_t ws_size,
                              hipStream_t stream) {
  (void)in_sizes; (void)n_in; (void)out_size; (void)ws_size;
  const float* feat[4] = {(const float*)d_in[0], (const float*)d_in[3], (const float*)d_in[6], (const float*)d_in[9]};
  const float* Wfc[4]  = {(const float*)d_in[1], (const float*)d_in[4], (const float*)d_in[7], (const float*)d_in[10]};
  const float* bfc[4]  = {(const float*)d_in[2], (const float*)d_in[5], (const float*)d_in[8], (const float*)d_in[11]};
  const float* type_emb = (const float*)d_in[12];
  const float* Wgcn = (const float*)d_in[13];
  const float* bgcn = (const float*)d_in[14];
  const float* Wre  = (const float*)d_in[15];
  const float* bre  = (const float*)d_in[16];
  const float* wtre = (const float*)d_in[17];
  const float* Wl   = (const float*)d_in[18];
  const float* Wr   = (const float*)d_in[19];
  const float* al   = (const float*)d_in[20];
  const float* ar   = (const float*)d_in[21];
  const float* Wrs  = (const float*)d_in[22];
  const float* Wrt  = (const float*)d_in[23];
  const float* Wfin = (const float*)d_in[24];
  const float* ln_g = (const float*)d_in[25];
  const float* ln_b = (const float*)d_in[26];
  const int* src = (const int*)d_in[27];
  const int* dst = (const int*)d_in[28];
  const int* node_type = (const int*)d_in[29];
  const int* seqs = (const int*)d_in[30];
  float* out = (float*)d_out;

  // workspace carve (floats)
  float* ws = (float*)d_ws;
  size_t o = 0;
  float* gh   = ws + o; o += (size_t)NN * EMB;
  float* xbuf = ws + o; o += (size_t)NN * EMB;
  float* nout = ws + o; o += NN;
  float* nin  = ws + o; o += NN;
  float* R    = ws + o; o += (size_t)NN * NTY;
  float* Rtmp = ws + o; o += (size_t)NN * NTY;
  float* h    = ws + o; o += (size_t)NN * SQ * EMB;
  half_t* wbf = (half_t*)(ws + o);   // 8 matrices x 65536 halves
  const int WM = EMB * EMB;          // 65536
  half_t* Wgcn_t = wbf;              // [2]
  half_t* Wl_t   = wbf + 2 * WM;     // [2]
  half_t* Wr_t   = wbf + 4 * WM;     // [2]
  half_t* Wfin_t = wbf + 6 * WM;     // [2]

  static const int rows[4]  = {4096, 2048, 1024, 1024};
  static const int dims[4]  = {128, 64, 32, 16};
  static const int offs[4]  = {0, 4096, 6144, 7168};

  // 1) per-type FC -> gh
  for (int i = 0; i < 4; ++i)
    k_fc<<<(rows[i] * EMB) / 256, 256, 0, stream>>>(feat[i], Wfc[i], bfc[i],
                                                    gh + (size_t)offs[i] * EMB, rows[i], dims[i]);

  // 2) degrees + R init
  k_node_init<<<NN / 256, 256, 0, stream>>>(nout, nin, R, type_emb, node_type);
  k_deg_edges<<<NE / 256, 256, 0, stream>>>(src, dst, nout, nin);
  k_deg_fin<<<NN / 256, 256, 0, stream>>>(nout);
  k_deg_fin<<<NN / 256, 256, 0, stream>>>(nin);

  // 3) weight convert+transpose to f16
  for (int k = 0; k < 2; ++k) {
    k_w2h_t<<<WM / 256, 256, 0, stream>>>(Wgcn + (size_t)k * WM, Wgcn_t + (size_t)k * WM);
    k_w2h_t<<<WM / 256, 256, 0, stream>>>(Wl   + (size_t)k * WM, Wl_t   + (size_t)k * WM);
    k_w2h_t<<<WM / 256, 256, 0, stream>>>(Wr   + (size_t)k * WM, Wr_t   + (size_t)k * WM);
    k_w2h_t<<<WM / 256, 256, 0, stream>>>(Wfin + (size_t)k * WM, Wfin_t + (size_t)k * WM);
  }

  // 4) output slab 0: l2n of PRE-GCN gh (reference appends ret[0] before the K loop)
  k_l2n_out<<<NN, 256, 0, stream>>>(gh, out);

  // 5) K=2 GCN + REConv layers
  for (int k = 0; k < 2; ++k) {
    k_gemm_scaled<<<NN / 32, 256, 0, stream>>>(gh, nout, Wgcn_t + (size_t)k * WM, xbuf);
    k_zero<<<(NN * EMB) / 256, 256, 0, stream>>>(gh, NN * EMB);
    k_scatter_emb<<<NE, 256, 0, stream>>>(xbuf, src, dst, gh);
    k_gcn_fin<<<(NN * EMB) / 256, 256, 0, stream>>>(gh, xbuf, nin, bgcn + (size_t)k * EMB);

    k_re_msg<<<(NN * NTY) / 256, 256, 0, stream>>>(R, nout, Wre + (size_t)k * NTY * NTY,
                                                   wtre + (size_t)k * NTY, node_type, Rtmp);
    k_zero<<<(NN * NTY) / 256, 256, 0, stream>>>(R, NN * NTY);
    k_re_scatter<<<(NE * NTY) / 256, 256, 0, stream>>>(Rtmp, src, dst, R);
    k_re_fin<<<(NN * NTY) / 256, 256, 0, stream>>>(R, Rtmp, nin, bre + (size_t)k * NTY);
  }

  // 6) gather h = gh[seqs]
  k_gather<<<((size_t)NN * SQ * EMB) / 256, 256, 0, stream>>>(gh, seqs, h);

  // 7) L=2 attention layers (dynamic LDS ~123 KB per workgroup; gfx1250 allows up to 320 KB)
  size_t smem = 32 * ASTR * sizeof(half_t)          // f16 tile
              + 2 * 32 * FSTR * sizeof(float)       // fl, fr
              + NHD * SQ * SQ * sizeof(float)       // scores
              + 2 * SQ * NHD * NTY * sizeof(float); // rk, rq
  for (int l = 0; l < 2; ++l) {
    k_attn<<<NN, 256, smem, stream>>>(h, R, seqs,
                                      Wl_t + (size_t)l * WM, Wr_t + (size_t)l * WM, Wfin_t + (size_t)l * WM,
                                      al + (size_t)l * HD, ar + (size_t)l * HD,
                                      Wrs + (size_t)l * NTY * 32, Wrt + (size_t)l * NTY * 32,
                                      ln_g + (size_t)l * EMB, ln_b + (size_t)l * EMB,
                                      out, EMB * (l + 1));
  }
}
